// CTC_309237645475
// MI455X (gfx1250) — matrix operand order repaired and checked by reading, not compiled, not run on hardware
//
#include <hip/hip_runtime.h>
#include <hip/hip_bf16.h>
#include <math.h>

typedef __attribute__((ext_vector_type(16))) __bf16 v16bf;
typedef __attribute__((ext_vector_type(8)))  __bf16 v8bf;
typedef __attribute__((ext_vector_type(8)))  float  v8f;

#define B_    16
#define T_    1000
#define D_    512
#define V_    5000
#define LMAX_ 120
#define S_    (2*LMAX_+1)   // 241 extended states
#define NL_   128           // padded label columns (121 real)
#define NEGV  (-1.0e9f)
#define KSTEPS (D_/32)      // 16 WMMA K-steps
#define ROWSPAN 520         // bf16 elems per LDS row (1040 B; bank-conflict padding)

// ---------------- fp32 -> bf16 pre-conversion ----------------
__global__ void cvt_f32_bf16(const float* __restrict__ in, __bf16* __restrict__ out, int n) {
  int i = blockIdx.x * blockDim.x + threadIdx.x;
  if (i < n) out[i] = (__bf16)in[i];
}

__global__ void zero_out(float* o) { *o = 0.0f; }

// Load this lane's A fragment for all 16 K-steps into registers (128 VGPRs).
__device__ __forceinline__ void load_a_frags(const __bf16* __restrict__ arow,
                                             int aoff, v16bf* areg) {
  #pragma unroll
  for (int k = 0; k < KSTEPS; ++k) {
    v8bf alo = *(const v8bf*)(arow + k * 32 + aoff);
    v8bf ahi = *(const v8bf*)(arow + k * 32 + 8 + aoff);
    #pragma unroll
    for (int i = 0; i < 8; ++i) { areg[k][i] = alo[i]; areg[k][8 + i] = ahi[i]; }
  }
}

// Async global->LDS copy of one 32-column pair of W tiles (32 rows x 1KB),
// spread across 128 threads (16 x b128 per thread). Tracked by ASYNCcnt.
__device__ __forceinline__ void issue_pair_copy(const __bf16* __restrict__ Wb,
                                                int ntp, __bf16* bufS, int tid) {
  #pragma unroll
  for (int i = 0; i < 16; ++i) {
    const int chunk = i * 128 + tid;        // 0..2047 16B-chunks
    const int row   = chunk >> 6;           // 0..31  (col = ntp*32 + row)
    const int offB  = (chunk & 63) * 16;    // byte offset within 1KB row
    int col  = ntp * 32 + row;
    int colc = (col < V_) ? col : 0;        // clamp pad columns
    const uint64_t ga = (uint64_t)(uintptr_t)((const char*)(Wb + (size_t)colc * D_) + offB);
    const uint32_t la = (uint32_t)(uintptr_t)((char*)bufS + row * (ROWSPAN * 2) + offB);
    asm volatile("global_load_async_to_lds_b128 %0, %1, off"
                 :: "v"(la), "v"(ga) : "memory");
  }
}

// ---------------- Kernel 1: fused GEMM + streaming logsumexp over V ----------------
// A register-resident per wave; B double-buffered in LDS via async copies shared
// by all 4 waves; two accumulator chains; branchless streaming (max,sumexp).
__global__ __launch_bounds__(128)
void logz_kernel(const __bf16* __restrict__ A,   // (16000, 512) bf16
                 const __bf16* __restrict__ Wb,  // (5000, 512)  bf16
                 const float*  __restrict__ bias,
                 float*        __restrict__ logZ) // (16000)
{
  __shared__ __bf16 bstage[2][32 * ROWSPAN];     // 66,560 B

  const int tid  = threadIdx.x;
  const int lane = tid & 31;
  const int wave = tid >> 5;
  const int hi   = lane >> 4;
  const int l15  = lane & 15;
  const int rowBase = blockIdx.x * 64 + wave * 16;

  const int aoff = hi ? 16 : 0;
  const int eoff = hi ? 16 : 0;   // B fragment elem offset (32 B) for hi lanes

  v16bf areg[KSTEPS];
  load_a_frags(A + (size_t)(rowBase + l15) * D_, aoff, areg);

  float m[8], s[8];
  #pragma unroll
  for (int e = 0; e < 8; ++e) { m[e] = -3.0e38f; s[e] = 0.0f; }

  const int NTP = (V_ + 31) / 32;  // 157 tile-pairs (314 tiles; pads masked)
  issue_pair_copy(Wb, 0, &bstage[0][0], tid);

  for (int ntp = 0; ntp < NTP; ++ntp) {
    const int cur = ntp & 1;
    if (ntp + 1 < NTP) {
      issue_pair_copy(Wb, ntp + 1, &bstage[cur ^ 1][0], tid);
      asm volatile("s_wait_asynccnt 0x10" ::: "memory");  // prev pair done
    } else {
      asm volatile("s_wait_asynccnt 0x0" ::: "memory");
    }
    __syncthreads();                                      // publish to all waves

    // fragment base pointers in LDS (tile0: rows 0..15, tile1: rows 16..31)
    const __bf16* f0 = &bstage[cur][0] + l15 * ROWSPAN + eoff;
    const __bf16* f1 = f0 + 16 * ROWSPAN;

    v8f acc0 = {}, acc1 = {};
    v16bf b0 = *(const v16bf*)(f0);
    v16bf b1 = *(const v16bf*)(f1);
    #pragma unroll
    for (int k = 0; k < KSTEPS; ++k) {
      v16bf n0, n1;
      if (k + 1 < KSTEPS) {                    // 2-stage pipeline: prefetch k+1
        n0 = *(const v16bf*)(f0 + (k + 1) * 32);
        n1 = *(const v16bf*)(f1 + (k + 1) * 32);
      }
      acc0 = __builtin_amdgcn_wmma_f32_16x16x32_bf16(
                 false, areg[k], false, b0, (short)0, acc0, false, false);
      acc1 = __builtin_amdgcn_wmma_f32_16x16x32_bf16(
                 false, areg[k], false, b1, (short)0, acc1, false, false);
      b0 = n0; b1 = n1;
    }

    // branchless streaming logsumexp; pad columns contribute exp(-huge)=0
    const int col0  = ntp * 32 + l15;
    const int col1  = col0 + 16;
    const int colc0 = (col0 < V_) ? col0 : 0;
    const int colc1 = (col1 < V_) ? col1 : 0;
    const float bb0 = bias[colc0] + ((col0 < V_) ? 0.0f : -1.0e38f);
    const float bb1 = bias[colc1] + ((col1 < V_) ? 0.0f : -1.0e38f);
    #pragma unroll
    for (int e = 0; e < 8; ++e) {
      float x0   = acc0[e] + bb0;
      float x1   = acc1[e] + bb1;
      float xm   = fmaxf(x0, x1);
      float mnew = fmaxf(m[e], xm);
      s[e] = s[e] * __expf(m[e] - mnew)
           + __expf(x0 - mnew) + __expf(x1 - mnew);
      m[e] = mnew;
    }
    __syncthreads();   // all waves done reading before buffer is overwritten
  }

  // combine (m,s) across the 16 lanes sharing each row (stays within half-wave)
  #pragma unroll
  for (int mask = 1; mask < 16; mask <<= 1) {
    #pragma unroll
    for (int e = 0; e < 8; ++e) {
      float om = __shfl_xor(m[e], mask, 32);
      float os = __shfl_xor(s[e], mask, 32);
      float M  = fmaxf(m[e], om);
      s[e] = s[e] * __expf(m[e] - M) + os * __expf(om - M);
      m[e] = M;
    }
  }
  if (l15 == 0) {
    const int rb = rowBase + (hi ? 8 : 0);
    #pragma unroll
    for (int e = 0; e < 8; ++e) logZ[rb + e] = m[e] + __logf(s[e]);
  }
}

// ---------------- Kernel 2: label-column logits (gathered GEMM, 2 GFLOP) ----------------
__global__ __launch_bounds__(128)
void lablogit_kernel(const __bf16* __restrict__ A,
                     const __bf16* __restrict__ Wb,
                     const float*  __restrict__ bias,
                     const int*    __restrict__ ys,   // (B, LMAX)
                     float*        __restrict__ lab)  // (B, T, NL_)
{
  const int lane = threadIdx.x & 31;
  const int wave = threadIdx.x >> 5;
  const int hi   = lane >> 4;
  const int l15  = lane & 15;
  const int b    = blockIdx.y;
  const int tBase = blockIdx.x * 64 + wave * 16;

  int tA = tBase + l15; if (tA >= T_) tA = T_ - 1;   // clamp loads; stores guarded
  const int aoff = hi ? 16 : 0;
  const int boff = hi ? 16 : 0;

  v16bf areg[KSTEPS];
  load_a_frags(A + (size_t)(b * T_ + tA) * D_, aoff, areg);

  const int tRow = tBase + (hi ? 8 : 0);

  for (int ntp = 0; ntp < NL_ / 32; ++ntp) {          // 4 tile-pairs
    const int j0 = ntp * 32 + l15;
    const int j1 = j0 + 16;
    int lab0 = 0, lab1 = 0;
    if (j0 >= 1 && j0 <= LMAX_) lab0 = ys[b * LMAX_ + j0 - 1];
    if (j1 >= 1 && j1 <= LMAX_) lab1 = ys[b * LMAX_ + j1 - 1];
    const __bf16* wrow0 = Wb + (size_t)lab0 * D_ + boff;
    const __bf16* wrow1 = Wb + (size_t)lab1 * D_ + boff;

    v8f acc0 = {}, acc1 = {};
    #pragma unroll
    for (int k = 0; k < KSTEPS; ++k) {
      v16bf bv0 = *(const v16bf*)(wrow0 + k * 32);
      v16bf bv1 = *(const v16bf*)(wrow1 + k * 32);
      acc0 = __builtin_amdgcn_wmma_f32_16x16x32_bf16(
                 false, areg[k], false, bv0, (short)0, acc0, false, false);
      acc1 = __builtin_amdgcn_wmma_f32_16x16x32_bf16(
                 false, areg[k], false, bv1, (short)0, acc1, false, false);
    }

    const float bb0 = bias[lab0];
    const float bb1 = bias[lab1];
    #pragma unroll
    for (int e = 0; e < 8; ++e) {
      if (tRow + e < T_) {
        float* dst = lab + (size_t)(b * T_ + tRow + e) * NL_;
        dst[j0] = acc0[e] + bb0;
        dst[j1] = acc1[e] + bb1;
      }
    }
  }
}

// ---------------- Kernel 3: sequential CTC forward scan ----------------
__global__ __launch_bounds__(256)
void ctc_kernel(const float* __restrict__ lab,   // (B,T,NL_)
                const float* __restrict__ logZ,  // (B*T)
                const int*   __restrict__ ys,
                const int*   __restrict__ hlens,
                const int*   __restrict__ ylens,
                float*       __restrict__ out)
{
  __shared__ float alpha[S_];
  const int b = blockIdx.x;
  const int s = threadIdx.x;
  const bool valid = (s < S_);

  int  jOf = 0;
  bool skipOK = false;
  if (valid && (s & 1)) {
    jOf = (s + 1) >> 1;                          // odd s -> label ys[(s-1)/2]
    if (s >= 3)
      skipOK = (ys[b * LMAX_ + ((s - 1) >> 1)] != ys[b * LMAX_ + ((s - 1) >> 1) - 1]);
  }
  const int Tend = hlens[b];                     // alpha frozen past hlens
  const float* labB = lab  + (size_t)b * T_ * NL_;
  const float* lzB  = logZ + b * T_;

  if (valid) {
    const float lp0 = labB[jOf] - lzB[0];
    alpha[s] = (s <= 1) ? lp0 : NEGV;
  }
  __syncthreads();

  for (int t = 1; t < Tend; ++t) {
    float a0 = NEGV, a1 = NEGV, a2 = NEGV, lp = 0.0f;
    if (valid) {
      a0 = alpha[s];
      if (s >= 1)  a1 = alpha[s - 1];
      if (skipOK)  a2 = alpha[s - 2];
      lp = labB[(size_t)t * NL_ + jOf] - lzB[t];
    }
    __syncthreads();
    if (valid) {
      float mm = fmaxf(a0, fmaxf(a1, a2));
      float l  = mm + __logf(__expf(a0 - mm) + __expf(a1 - mm) + __expf(a2 - mm));
      alpha[s] = l + lp;
    }
    __syncthreads();
  }

  if (s == 0) {
    const int end = 2 * ylens[b];
    const float al = alpha[end], ap = alpha[end - 1];
    const float mm = fmaxf(al, ap);
    const float loss = -(mm + __logf(__expf(al - mm) + __expf(ap - mm)));
    atomicAdd(out, loss / (float)B_);
  }
}

// ---------------- host-side launch ----------------
extern "C" void kernel_launch(void* const* d_in, const int* in_sizes, int n_in,
                              void* d_out, int out_size, void* d_ws, size_t ws_size,
                              hipStream_t stream) {
  const float* hs    = (const float*)d_in[0];  // (B,T,D)
  const int*   hlens = (const int*)  d_in[1];  // (B)
  const int*   ys    = (const int*)  d_in[2];  // (B,LMAX)
  const int*   ylens = (const int*)  d_in[3];  // (B)
  const float* W     = (const float*)d_in[4];  // (V,D)
  const float* bias  = (const float*)d_in[5];  // (V)
  float*       out   = (float*)d_out;

  // workspace layout (all offsets 32B aligned)
  char* ws = (char*)d_ws;
  __bf16* Abf  = (__bf16*)(ws);                               // 16,384,000 B
  __bf16* Wbf  = (__bf16*)(ws + 16384000);                    //  5,120,000 B
  float*  labL = (float*) (ws + 16384000 + 5120000);          //  8,192,000 B
  float*  logZ = (float*) (ws + 16384000 + 5120000 + 8192000);//     64,000 B

  const int nA = B_ * T_ * D_;   // 8,192,000
  const int nW = V_ * D_;        // 2,560,000
  cvt_f32_bf16<<<(nA + 255) / 256, 256, 0, stream>>>(hs, Abf, nA);
  cvt_f32_bf16<<<(nW + 255) / 256, 256, 0, stream>>>(W,  Wbf, nW);

  logz_kernel<<<(B_ * T_) / 64, 128, 0, stream>>>(Abf, Wbf, bias, logZ);
  lablogit_kernel<<<dim3(16, B_), 128, 0, stream>>>(Abf, Wbf, bias, ys, labL);

  zero_out<<<1, 1, 0, stream>>>(out);
  ctc_kernel<<<B_, 256, 0, stream>>>(labL, logZ, ys, hlens, ylens, out);
}
